// CBSA_45389214384209
// MI455X (gfx1250) — matrix operand order repaired and checked by reading, not compiled, not run on hardware
//
#include <hip/hip_runtime.h>

// ---------------- problem constants ----------------
#define BATCH  8
#define CDIM   256
#define HEADS  8
#define DH     64
#define INNER  512          // HEADS*DH
#define NTOK   4096         // 64*64 spatial tokens
#define BN     (BATCH*NTOK) // 32768 rows
#define SCALE_ 0.125f       // DH^-0.5

typedef __attribute__((ext_vector_type(16))) __bf16 v16bf;
typedef __attribute__((ext_vector_type(8)))  float  v8f;

static __device__ __forceinline__ v8f wmma_bf16(v16bf a, v16bf b, v8f c) {
  // D = A(16x32) * B(32x16) + C, f32 accumulate
  return __builtin_amdgcn_wmma_f32_16x16x32_bf16(false, a, false, b, (short)0, c, false, false);
}

// ---- CDNA5 async copy: global -> LDS, 16B per lane, ASYNCcnt tracked ----
static __device__ __forceinline__ void async_b128(unsigned lds_off, const void* gaddr) {
  asm volatile("global_load_async_to_lds_b128 %0, %1, off"
               :: "v"(lds_off), "v"(gaddr) : "memory");
}
static __device__ __forceinline__ void async_wait0() {
  asm volatile("s_wait_asynccnt 0" ::: "memory");
}
static __device__ __forceinline__ unsigned lds_off(const void* p) {
  return (unsigned)(unsigned long long)p;   // AS(3) pointers are LDS byte offsets
}

static __device__ __forceinline__ unsigned pack2bf(float a, float b) {
  union { __bf16 h[2]; unsigned u; } p;
  p.h[0] = (__bf16)a; p.h[1] = (__bf16)b;
  return p.u;
}

// A fragment: 16x32 bf16, source LDS tile row-major [row][k], pairs contiguous.
// lane: m = lane&15, hl = lane>>4.  K(e) halves: {0-15} / {16-31}.
static __device__ __forceinline__ v16bf fragA(const __bf16* base, int rs, int row, int kbase, int hl) {
  union { v16bf v; unsigned int u[8]; } f;
  const __bf16* rp = base + row * rs + kbase;
#pragma unroll
  for (int q8 = 0; q8 < 8; ++q8) {
    int k0 = 2 * q8 + 16 * hl;
    f.u[q8] = *(const unsigned int*)(rp + k0);
  }
  return f.v;
}

// B fragment: 32x16 bf16, n = lane&15, K halves {0-15}/{16-31}.
// Source LDS stored [n][K] row-major (pairs contiguous along K): u32 gathers.
static __device__ __forceinline__ v16bf fragB_nk(const __bf16* base, int rs, int nrow, int kbase, int hl) {
  union { v16bf v; unsigned int u[8]; } f;
  const __bf16* rp = base + nrow * rs + kbase;
#pragma unroll
  for (int q8 = 0; q8 < 8; ++q8) {
    int k0 = 2 * q8 + 16 * hl;
    f.u[q8] = *(const unsigned int*)(rp + k0);
  }
  return f.v;
}

// B fragment from LDS stored [K][n] row-major (pairs strided): u16 gathers.
static __device__ __forceinline__ v16bf fragB_kn(const __bf16* base, int rs, int ncol, int kbase, int hl) {
  union { v16bf v; __bf16 h[16]; } f;
#pragma unroll
  for (int q8 = 0; q8 < 8; ++q8) {
    int k0 = 2 * q8 + 16 * hl + kbase;
    f.h[2 * q8]     = base[(k0)     * rs + ncol];
    f.h[2 * q8 + 1] = base[(k0 + 1) * rs + ncol];
  }
  return f.v;
}

#define ZERO8 {0.f,0.f,0.f,0.f,0.f,0.f,0.f,0.f}

// ============================================================
// K1: proj[b,n,inner] = x_seq[b,n,c] @ W_proj[c,inner]  (bf16 out)
//   A[row, c] = x[b, c, n]   (row = b*4096 + n)
//   grid = (4 Ntiles, 256 Mtiles), tile 128x128, K-step 32
//   Staging: batched f32 loads -> packed bf16-pair LDS stores.
//   Both LDS tiles are [row][k] so all gathers are u32.
// ============================================================
__global__ __launch_bounds__(256) void k_proj(const float* __restrict__ x,
                                              const float* __restrict__ Wp,
                                              __bf16* __restrict__ proj) {
  const int nt = blockIdx.x, mt = blockIdx.y;
  const int row0 = mt * 128;
  const int b  = row0 >> 12;
  const int n0 = row0 & (NTOK - 1);
  const int col0 = nt * 128;

  __shared__ __bf16 As[128][34];   // [m][k]
  __shared__ __bf16 Bs[128][34];   // [n][k]  (B stored n-major -> u32 gathers)

  const int tid = threadIdx.x, lane = tid & 31, wv = tid >> 5;
  const int hl = lane >> 4, lm = lane & 15;
  const int wm = (wv & 1) * 64;   // wave M base (2 blocks of 64)
  const int wn = (wv >> 1) * 32;  // wave N base (4 blocks of 32)

  const int mloc = tid & 127;         // staging row
  const int kgrp = (tid >> 7) * 16;   // 16 k's per thread group

  v8f acc[4][2];
#pragma unroll
  for (int i = 0; i < 4; ++i) { v8f z = ZERO8; acc[i][0] = z; acc[i][1] = z; }

  for (int kt = 0; kt < CDIM; kt += 32) {
    float rA[16], rB[16];
#pragma unroll
    for (int j = 0; j < 16; ++j)
      rA[j] = x[(size_t)(b * CDIM + kt + kgrp + j) * NTOK + n0 + mloc];
#pragma unroll
    for (int j = 0; j < 16; ++j)
      rB[j] = Wp[(size_t)(kt + kgrp + j) * INNER + col0 + mloc];
#pragma unroll
    for (int jj = 0; jj < 8; ++jj) {
      *(unsigned*)&As[mloc][kgrp + 2 * jj] = pack2bf(rA[2 * jj], rA[2 * jj + 1]);
      *(unsigned*)&Bs[mloc][kgrp + 2 * jj] = pack2bf(rB[2 * jj], rB[2 * jj + 1]);
    }
    __syncthreads();
    v16bf af[4], bf[2];
#pragma unroll
    for (int m4 = 0; m4 < 4; ++m4) af[m4] = fragA(&As[0][0], 34, wm + m4 * 16 + lm, 0, hl);
#pragma unroll
    for (int n2 = 0; n2 < 2; ++n2) bf[n2] = fragB_nk(&Bs[0][0], 34, wn + n2 * 16 + lm, 0, hl);
#pragma unroll
    for (int m4 = 0; m4 < 4; ++m4)
#pragma unroll
      for (int n2 = 0; n2 < 2; ++n2) acc[m4][n2] = wmma_bf16(af[m4], bf[n2], acc[m4][n2]);
    __syncthreads();
  }
#pragma unroll
  for (int m4 = 0; m4 < 4; ++m4)
#pragma unroll
    for (int n2 = 0; n2 < 2; ++n2) {
      int col = col0 + wn + n2 * 16 + lm;
#pragma unroll
      for (int r = 0; r < 8; ++r) {
        int m = wm + m4 * 16 + r + 8 * hl;
        proj[(size_t)(row0 + m) * INNER + col] = (__bf16)acc[m4][n2][r];
      }
    }
}

// ============================================================
// K2: 8x8 block-mean pooling -> rep f32 [b,64,inner]
//     and pre-scaled bf16 queries q[bh,64,dh]
//     Coalesced: 2 channels per thread, loop over 64 positions.
// ============================================================
__global__ __launch_bounds__(256) void k_pool(const __bf16* __restrict__ proj,
                                              float* __restrict__ rep,
                                              __bf16* __restrict__ qbuf) {
  const int bg = blockIdx.x, b = bg >> 6, g = bg & 63;
  const int gy = g >> 3, gx = g & 7;
  const int c0 = threadIdx.x * 2;
  float s0 = 0.f, s1 = 0.f;
#pragma unroll 4
  for (int iy = 0; iy < 8; ++iy)
    for (int ix = 0; ix < 8; ++ix) {
      int n = (gy * 8 + iy) * 64 + gx * 8 + ix;
      const __bf16* rp = &proj[(size_t)(b * NTOK + n) * INNER + c0];
      __builtin_prefetch(rp + INNER, 0, 1);
      union { unsigned u; __bf16 h[2]; } p;
      p.u = *(const unsigned*)rp;
      s0 += (float)p.h[0];
      s1 += (float)p.h[1];
    }
  s0 *= (1.f / 64.f); s1 *= (1.f / 64.f);
  rep[(size_t)(b * 64 + g) * INNER + c0]     = s0;
  rep[(size_t)(b * 64 + g) * INNER + c0 + 1] = s1;
  {
    int h = c0 >> 6, d = c0 & 63;   // c0 even -> both channels in same head row
    unsigned qp = pack2bf(s0 * SCALE_, s1 * SCALE_);
    *(unsigned*)&qbuf[(size_t)((b * HEADS + h) * 64 + g) * DH + d] = qp;
  }
}

// ============================================================
// K3: per (b,h): flash attention (64 queries x 4096 keys) with
//     raw scores spilled to f32 ws, then rep2 residual + stage-2
//     64x64 attention, all WMMA.  K/V chunks moved with
//     global_load_async_to_lds_b128.  grid = 64 blocks.
// ============================================================
__global__ __launch_bounds__(256) void k_attn(const __bf16* __restrict__ proj,
                                              const __bf16* __restrict__ qbuf,
                                              const float* __restrict__ rep,
                                              const float* __restrict__ step_rep,
                                              const float* __restrict__ step_x,
                                              float* __restrict__ dots,
                                              float* __restrict__ stats,
                                              __bf16* __restrict__ ybuf) {
  const int bh = blockIdx.x, b = bh >> 3, h = bh & 7;
  __shared__ __bf16 q_s[64][72];     // async target: 144B rows (16B aligned)
  __shared__ __bf16 k_s[64][72];     // K/V chunk (same data), async target
  __shared__ float  s_s[64][68];     // scores f32 (reused for stage-2)
  __shared__ __bf16 p_s[64][68];     // exp() probs bf16 (reused for stage-2)
  __shared__ __bf16 r2_s[64][66];    // rep2 bf16
  __shared__ __bf16 r2q_s[64][66];   // rep2*SCALE bf16
  __shared__ float mx_s[64], sum_s[64], al_s[64];

  const int tid = threadIdx.x, lane = tid & 31, wv = tid >> 5;
  const int hl = lane >> 4, lm = lane & 15;
  const int wm = (wv >> 1) * 16;     // this wave's 16 query rows
  const int wd = (wv & 1) * 32;      // this wave's two 16-wide d tiles

  // async-load q tile (64 rows x 64 bf16 = 8 chunks/row)
#pragma unroll
  for (int j = 0; j < 2; ++j) {
    int cid = tid + j * 256;
    int row = cid >> 3, cm = cid & 7;
    async_b128(lds_off(&q_s[row][cm * 8]),
               &qbuf[(size_t)(bh * 64 + row) * DH + cm * 8]);
  }
  if (tid < 64) { mx_s[tid] = -1e30f; sum_s[tid] = 0.f; }
  async_wait0();
  __syncthreads();

  v16bf qa0 = fragA(&q_s[0][0], 72, wm + lm, 0,  hl);
  v16bf qa1 = fragA(&q_s[0][0], 72, wm + lm, 32, hl);
  v8f oacc[2]; { v8f z = ZERO8; oacc[0] = z; oacc[1] = z; }

  for (int c0 = 0; c0 < NTOK; c0 += 64) {
    // async-load K/V chunk
#pragma unroll
    for (int j = 0; j < 2; ++j) {
      int cid = tid + j * 256;
      int row = cid >> 3, cm = cid & 7;
      async_b128(lds_off(&k_s[row][cm * 8]),
                 &proj[(size_t)(b * NTOK + c0 + row) * INNER + h * DH + cm * 8]);
    }
    async_wait0();
    __syncthreads();
    // scores: q @ K^T  (each wave: 2 key tiles)
#pragma unroll
    for (int tt = 0; tt < 2; ++tt) {
      int nb = ((wv & 1) * 2 + tt) * 16;
      v16bf b0 = fragB_nk(&k_s[0][0], 72, nb + lm, 0,  hl);
      v16bf b1 = fragB_nk(&k_s[0][0], 72, nb + lm, 32, hl);
      v8f d0 = ZERO8;
      d0 = wmma_bf16(qa0, b0, d0);
      d0 = wmma_bf16(qa1, b1, d0);
#pragma unroll
      for (int r = 0; r < 8; ++r) s_s[wm + r + 8 * hl][nb + lm] = d0[r];
    }
    __syncthreads();
    // spill raw scores (pass-2 input), coalesced
    for (int i = tid; i < 64 * 64; i += 256) {
      int m = i >> 6, kk = i & 63;
      dots[(size_t)(bh * 64 + m) * NTOK + c0 + kk] = s_s[m][kk];
    }
    // online softmax stats, one row per thread
    if (tid < 64) {
      int m = tid;
      float cmax = -1e30f;
      for (int kk = 0; kk < 64; ++kk) cmax = fmaxf(cmax, s_s[m][kk]);
      float nm = fmaxf(mx_s[m], cmax);
      float al = __expf(mx_s[m] - nm);
      float s = 0.f;
      for (int kk = 0; kk < 64; ++kk) {
        float e = __expf(s_s[m][kk] - nm);
        p_s[m][kk] = (__bf16)e;
        s += e;
      }
      sum_s[m] = sum_s[m] * al + s;
      mx_s[m] = nm; al_s[m] = al;
    }
    __syncthreads();
    // rescale accumulators, then P @ V
#pragma unroll
    for (int tt = 0; tt < 2; ++tt)
#pragma unroll
      for (int r = 0; r < 8; ++r) oacc[tt][r] *= al_s[wm + r + 8 * hl];
#pragma unroll
    for (int ks = 0; ks < 2; ++ks) {
      v16bf pa = fragA(&p_s[0][0], 68, wm + lm, ks * 32, hl);
#pragma unroll
      for (int tt = 0; tt < 2; ++tt) {
        v16bf vb = fragB_kn(&k_s[0][0], 72, wd + tt * 16 + lm, ks * 32, hl);
        oacc[tt] = wmma_bf16(pa, vb, oacc[tt]);
      }
    }
    __syncthreads();
  }

  // finalize stage-1: rep2 = rep + step_rep * (acc / sum)
  const float srh = step_rep[h];
#pragma unroll
  for (int tt = 0; tt < 2; ++tt)
#pragma unroll
    for (int r = 0; r < 8; ++r) {
      int m = wm + r + 8 * hl, d = wd + tt * 16 + lm;
      float od = oacc[tt][r] / sum_s[m];
      float r2 = rep[(size_t)(b * 64 + m) * INNER + h * DH + d] + srh * od;
      r2_s[m][d]  = (__bf16)r2;
      r2q_s[m][d] = (__bf16)(r2 * SCALE_);
    }
  if (tid < 64) {
    stats[(size_t)(bh * 64 + tid) * 2 + 0] = mx_s[tid];
    stats[(size_t)(bh * 64 + tid) * 2 + 1] = 1.f / sum_s[tid];
  }
  __syncthreads();

  // stage 2: dots2 = (rep2*SCALE) @ rep2^T
  {
    v16bf a0 = fragA(&r2q_s[0][0], 66, wm + lm, 0,  hl);
    v16bf a1 = fragA(&r2q_s[0][0], 66, wm + lm, 32, hl);
#pragma unroll
    for (int tt = 0; tt < 2; ++tt) {
      int nb = ((wv & 1) * 2 + tt) * 16;
      v16bf b0 = fragB_nk(&r2_s[0][0], 66, nb + lm, 0,  hl);
      v16bf b1 = fragB_nk(&r2_s[0][0], 66, nb + lm, 32, hl);
      v8f d0 = ZERO8;
      d0 = wmma_bf16(a0, b0, d0);
      d0 = wmma_bf16(a1, b1, d0);
#pragma unroll
      for (int r = 0; r < 8; ++r) s_s[wm + r + 8 * hl][nb + lm] = d0[r];
    }
  }
  __syncthreads();
  if (tid < 64) {            // exact softmax over 64 tokens
    int m = tid;
    float cmax = -1e30f;
    for (int kk = 0; kk < 64; ++kk) cmax = fmaxf(cmax, s_s[m][kk]);
    float s = 0.f;
    for (int kk = 0; kk < 64; ++kk) s += __expf(s_s[m][kk] - cmax);
    float inv = 1.f / s;
    for (int kk = 0; kk < 64; ++kk)
      p_s[m][kk] = (__bf16)(__expf(s_s[m][kk] - cmax) * inv);
  }
  __syncthreads();
  // y = attn2 @ rep2, scaled by step_x
  {
    v8f yac[2]; { v8f z = ZERO8; yac[0] = z; yac[1] = z; }
#pragma unroll
    for (int ks = 0; ks < 2; ++ks) {
      v16bf pa = fragA(&p_s[0][0], 68, wm + lm, ks * 32, hl);
#pragma unroll
      for (int tt = 0; tt < 2; ++tt) {
        v16bf vb = fragB_kn(&r2_s[0][0], 66, wd + tt * 16 + lm, ks * 32, hl);
        yac[tt] = wmma_bf16(pa, vb, yac[tt]);
      }
    }
    const float sxh = step_x[h];
#pragma unroll
    for (int tt = 0; tt < 2; ++tt)
#pragma unroll
      for (int r = 0; r < 8; ++r) {
        int m = wm + r + 8 * hl, d = wd + tt * 16 + lm;
        ybuf[(size_t)(bh * 64 + m) * DH + d] = (__bf16)(yac[tt][r] * sxh);
      }
  }
}

// ============================================================
// K4: x_delta[b, n, h*64+d] = sum_m attn[m,n] * y[m,d]
//     attn regenerated from spilled scores + row stats.
//     y tile async-loaded.  grid = (16 key-chunks of 256, 64 bh)
// ============================================================
__global__ __launch_bounds__(256) void k_backproj(const float* __restrict__ dots,
                                                  const float* __restrict__ stats,
                                                  const __bf16* __restrict__ ybuf,
                                                  __bf16* __restrict__ xd) {
  const int bh = blockIdx.y, b = bh >> 3, h = bh & 7;
  const int n0 = blockIdx.x * 256;
  __shared__ __bf16 pT[256][66];     // attn^T chunk [n_local][m]
  __shared__ __bf16 y_s[64][72];     // async target
  __shared__ float mx_l[64], is_l[64];

  const int tid = threadIdx.x, lane = tid & 31, wv = tid >> 5;
  const int hl = lane >> 4, lm = lane & 15;

#pragma unroll
  for (int j = 0; j < 2; ++j) {
    int cid = tid + j * 256;
    int row = cid >> 3, cm = cid & 7;
    async_b128(lds_off(&y_s[row][cm * 8]),
               &ybuf[(size_t)(bh * 64 + row) * DH + cm * 8]);
  }
  if (tid < 64) {
    mx_l[tid] = stats[(size_t)(bh * 64 + tid) * 2 + 0];
    is_l[tid] = stats[(size_t)(bh * 64 + tid) * 2 + 1];
  }
  async_wait0();
  __syncthreads();
  for (int i = tid; i < 64 * 256; i += 256) {            // coalesced score reads
    int m = i >> 8, nn = i & 255;
    float v = dots[(size_t)(bh * 64 + m) * NTOK + n0 + nn];
    pT[nn][m] = (__bf16)(__expf(v - mx_l[m]) * is_l[m]);
  }
  __syncthreads();

  v16bf bf[4][2];
#pragma unroll
  for (int dc = 0; dc < 4; ++dc)
#pragma unroll
    for (int ks = 0; ks < 2; ++ks)
      bf[dc][ks] = fragB_kn(&y_s[0][0], 72, dc * 16 + lm, ks * 32, hl);

#pragma unroll
  for (int nt2 = 0; nt2 < 2; ++nt2) {
    int nb = (wv * 2 + nt2) * 16;
    v16bf a0 = fragA(&pT[0][0], 66, nb + lm, 0,  hl);
    v16bf a1 = fragA(&pT[0][0], 66, nb + lm, 32, hl);
#pragma unroll
    for (int dc = 0; dc < 4; ++dc) {
      v8f acc = ZERO8;
      acc = wmma_bf16(a0, bf[dc][0], acc);
      acc = wmma_bf16(a1, bf[dc][1], acc);
#pragma unroll
      for (int r = 0; r < 8; ++r) {
        int n = n0 + nb + r + 8 * hl;
        xd[(size_t)(b * NTOK + n) * INNER + h * DH + dc * 16 + lm] = (__bf16)acc[r];
      }
    }
  }
}

// ============================================================
// K5: out[b,c,n] = x_delta[b,n,:] @ W_out[:,c] + b_out[c]
//     A tile async-loaded (bf16 ws), B tile batched-staged.
//     grid = (2 Ntiles, 256 Mtiles), tile 128x128, K-step 32
// ============================================================
__global__ __launch_bounds__(256) void k_out(const __bf16* __restrict__ xd,
                                             const float* __restrict__ Wo,
                                             const float* __restrict__ bias,
                                             float* __restrict__ out) {
  const int nt = blockIdx.x, mt = blockIdx.y;
  const int row0 = mt * 128;
  const int b  = row0 >> 12;
  const int n0 = row0 & (NTOK - 1);
  const int col0 = nt * 128;

  __shared__ __bf16 As[128][40];   // async target: 80B rows (16B aligned)
  __shared__ __bf16 Bs[128][34];   // [n][k] -> u32 gathers

  const int tid = threadIdx.x, lane = tid & 31, wv = tid >> 5;
  const int hl = lane >> 4, lm = lane & 15;
  const int wm = (wv & 1) * 64;
  const int wn = (wv >> 1) * 32;

  const int nloc2 = tid & 127;
  const int kgrp  = (tid >> 7) * 16;

  v8f acc[4][2];
#pragma unroll
  for (int i = 0; i < 4; ++i) { v8f z = ZERO8; acc[i][0] = z; acc[i][1] = z; }

  for (int kt = 0; kt < INNER; kt += 32) {
    // A: async bf16 copy (128 rows x 32 bf16 = 4 chunks/row)
#pragma unroll
    for (int j = 0; j < 2; ++j) {
      int cid = tid + j * 256;
      int row = cid >> 2, cm = cid & 3;
      async_b128(lds_off(&As[row][cm * 8]),
                 &xd[(size_t)(row0 + row) * INNER + kt + cm * 8]);
    }
    // B: batched f32 loads -> packed bf16 stores
    float rB[16];
#pragma unroll
    for (int j = 0; j < 16; ++j)
      rB[j] = Wo[(size_t)(kt + kgrp + j) * CDIM + col0 + nloc2];
#pragma unroll
    for (int jj = 0; jj < 8; ++jj)
      *(unsigned*)&Bs[nloc2][kgrp + 2 * jj] = pack2bf(rB[2 * jj], rB[2 * jj + 1]);
    async_wait0();
    __syncthreads();
    v16bf af[4], bfr[2];
#pragma unroll
    for (int m4 = 0; m4 < 4; ++m4) af[m4] = fragA(&As[0][0], 40, wm + m4 * 16 + lm, 0, hl);
#pragma unroll
    for (int n2 = 0; n2 < 2; ++n2) bfr[n2] = fragB_nk(&Bs[0][0], 34, wn + n2 * 16 + lm, 0, hl);
#pragma unroll
    for (int m4 = 0; m4 < 4; ++m4)
#pragma unroll
      for (int n2 = 0; n2 < 2; ++n2) acc[m4][n2] = wmma_bf16(af[m4], bfr[n2], acc[m4][n2]);
    __syncthreads();
  }
#pragma unroll
  for (int m4 = 0; m4 < 4; ++m4)
#pragma unroll
    for (int n2 = 0; n2 < 2; ++n2) {
      int col = col0 + wn + n2 * 16 + lm;
      float bb = bias[col];
#pragma unroll
      for (int r = 0; r < 8; ++r) {
        int nloc = n0 + wm + m4 * 16 + r + 8 * hl;
        out[(size_t)(b * CDIM + col) * NTOK + nloc] = acc[m4][n2][r] + bb;
      }
    }
}

// ============================================================
extern "C" void kernel_launch(void* const* d_in, const int* in_sizes, int n_in,
                              void* d_out, int out_size, void* d_ws, size_t ws_size,
                              hipStream_t stream) {
  (void)in_sizes; (void)n_in; (void)out_size; (void)ws_size;
  const float* x   = (const float*)d_in[0];
  const float* Wp  = (const float*)d_in[1];
  const float* srp = (const float*)d_in[2];
  const float* sx  = (const float*)d_in[3];
  const float* Wo  = (const float*)d_in[4];
  const float* bo  = (const float*)d_in[5];
  float* out = (float*)d_out;

  char* ws = (char*)d_ws;
  size_t o = 0;
  __bf16* proj  = (__bf16*)(ws + o); o += (size_t)BN * INNER * 2;                 // 33.5 MB
  __bf16* qbuf  = (__bf16*)(ws + o); o += (size_t)BATCH * HEADS * 64 * DH * 2;    // 0.5 MB
  float*  rep   = (float* )(ws + o); o += (size_t)BATCH * 64 * INNER * 4;         // 1 MB
  float*  dots  = (float* )(ws + o); o += (size_t)BATCH * HEADS * 64 * NTOK * 4;  // 67 MB
  __bf16* ybuf  = (__bf16*)(ws + o); o += (size_t)BATCH * HEADS * 64 * DH * 2;    // 0.5 MB
  float*  stats = (float* )(ws + o); o += (size_t)BATCH * HEADS * 64 * 2 * 4;     // 32 KB
  __bf16* xd    = (__bf16*)(ws + o); o += (size_t)BN * INNER * 2;                 // 33.5 MB

  k_proj    <<<dim3(4, 256),  256, 0, stream>>>(x, Wp, proj);
  k_pool    <<<dim3(512),     256, 0, stream>>>(proj, rep, qbuf);
  k_attn    <<<dim3(64),      256, 0, stream>>>(proj, qbuf, rep, srp, sx, dots, stats, ybuf);
  k_backproj<<<dim3(16, 64),  256, 0, stream>>>(dots, stats, ybuf, xd);
  k_out     <<<dim3(2, 256),  256, 0, stream>>>(xd, Wo, bo, out);
}
